// GraphNeuralNetworkGCP_76158360093390
// MI455X (gfx1250) — matrix operand order repaired and checked by reading, not compiled, not run on hardware
//
#include <hip/hip_runtime.h>
#include <hip/hip_bf16.h>
#include <math.h>

#define BB 8
#define NN 2048
#define CC 16
#define DD 64
#define TT 32
#define HH 100

typedef __attribute__((ext_vector_type(16))) _Float16 v16h;
typedef __attribute__((ext_vector_type(8)))  _Float16 v8h;
typedef __attribute__((ext_vector_type(4)))  _Float16 v4h;
typedef __attribute__((ext_vector_type(8)))  float    v8f;
typedef __attribute__((ext_vector_type(4)))  float    v4f;

__device__ __forceinline__ float sigm(float x) { return 1.f / (1.f + __expf(-x)); }

// ---------------------------------------------------------------- init state (fp32 + f16 shadow of vh)
__global__ void k_init(const float* __restrict__ ch_init, const float* __restrict__ v_init,
                       float* __restrict__ vh, float* __restrict__ vc,
                       float* __restrict__ ch, float* __restrict__ cc,
                       _Float16* __restrict__ vh16) {
  int i = blockIdx.x * blockDim.x + threadIdx.x;
  if (i < BB * NN * DD) {
    float v = v_init[i % DD];
    vh[i] = v; vc[i] = 0.f; vh16[i] = (_Float16)v;
  }
  if (i < BB * CC * DD) { ch[i] = ch_init[i]; cc[i] = 0.f; }
}

// ---------------------------------------------------------------- Mvv fp32 -> f16 (once; A stays L2-resident)
__global__ void k_convA(const float* __restrict__ Mvv, _Float16* __restrict__ Ah) {
  size_t i = (size_t)blockIdx.x * blockDim.x + threadIdx.x;   // over B*N*N/4
  if (i >= (size_t)BB * NN * NN / 4) return;
  v4f a = *(const v4f*)(Mvv + i * 4);
  v4h h = { (_Float16)a.x, (_Float16)a.y, (_Float16)a.z, (_Float16)a.w };
  *(v4h*)(Ah + i * 4) = h;
}

// ---------------------------------------------------------------- Wcat[col*192+k] = f16 concat [Wih | Whh]
__global__ void k_convW(const float* __restrict__ Wih, const float* __restrict__ Whh,
                        _Float16* __restrict__ Wcat) {
  int i = blockIdx.x * blockDim.x + threadIdx.x;   // over 256*192
  if (i >= 256 * 192) return;
  int col = i / 192, k = i % 192;
  float v = (k < 128) ? Wih[col * 128 + k] : Whh[col * 64 + (k - 128)];
  Wcat[i] = (_Float16)v;
}

// ---------------------------------------------------------------- c_msg MLP (per color row)
__global__ void k_cmsg(const float* __restrict__ ch,
                       const float* __restrict__ W1, const float* __restrict__ b1,
                       const float* __restrict__ W2, const float* __restrict__ b2,
                       float* __restrict__ cmsg) {
  __shared__ float row[DD];
  __shared__ float hid[HH];
  int bc = blockIdx.x;
  int t = threadIdx.x;              // 128 threads
  if (t < DD) row[t] = ch[bc * DD + t];
  __syncthreads();
  if (t < HH) {
    float a = b1[t];
    const float* w = W1 + t * DD;
    for (int k = 0; k < DD; ++k) a += row[k] * w[k];
    hid[t] = fmaxf(a, 0.f);
  }
  __syncthreads();
  if (t < DD) {
    float a = b2[t];
    const float* w = W2 + t * HH;
    for (int k = 0; k < HH; ++k) a += hid[k] * w[k];
    cmsg[bc * DD + t] = fmaxf(a, 0.f);
  }
}

// ---------------------------------------------------------------- csum[b,d] = sum_c mask * cmsg
__global__ void k_csum(const float* __restrict__ cmsg, const int* __restrict__ n_colors,
                       float* __restrict__ csum) {
  int b = blockIdx.x, d = threadIdx.x;   // 64 threads
  int nc = n_colors[b];
  float s = 0.f;
  for (int c = 0; c < CC; ++c) if (c < nc) s += cmsg[(b * CC + c) * DD + d];
  csum[b * DD + d] = s;
}

// ---------------------------------------------------------------- agg = (Mvv + I) @ vh   (WMMA f16, f16 A+B)
__global__ void __launch_bounds__(128) k_agg(const _Float16* __restrict__ Ah,
                                             const _Float16* __restrict__ vh16,
                                             const float* __restrict__ vh,
                                             float* __restrict__ agg) {
  __shared__ _Float16 Bs[DD * 32];          // transposed: [n*32 + k]
  int b = blockIdx.y;
  int wave = threadIdx.x >> 5;              // 4 waves per block
  int lane = threadIdx.x & 31;
  int lcol = lane & 15, lsel = lane >> 4;
  int m0 = blockIdx.x * 64 + wave * 16;     // 16-row tile per wave
  const _Float16* Ab   = Ah   + (size_t)b * NN * NN;
  const _Float16* vh16b = vh16 + (size_t)b * NN * DD;

  v8f acc0 = {}, acc1 = {}, acc2 = {}, acc3 = {};

  for (int k0 = 0; k0 < NN; k0 += 32) {
    __syncthreads();
    // stage vh16[k0:k0+32, 0:64] transposed into Bs[n*32+k]; v8h loads, b16 scatter stores
    for (int u = threadIdx.x; u < 32 * (DD / 8); u += 128) {   // 256 units
      int k = u >> 3, n8 = (u & 7) * 8;
      v8h v = *(const v8h*)&vh16b[(k0 + k) * DD + n8];
#pragma unroll
      for (int j = 0; j < 8; ++j) Bs[(n8 + j) * 32 + k] = v[j];
    }
    __syncthreads();
    // A fragment: row = m0+lcol, K = k0 + lsel*16 + {0..15}  (two 8-half loads)
    const _Float16* ap = Ab + (size_t)(m0 + lcol) * NN + k0 + lsel * 16;
    v8h alo = *(const v8h*)ap;
    v8h ahi = *(const v8h*)(ap + 8);
    v16h af = __builtin_shufflevector(alo, ahi, 0,1,2,3,4,5,6,7,8,9,10,11,12,13,14,15);
    // B fragments: col = tile*16 + lcol, 16 contiguous K starting at lsel*16
    v16h bf0 = *(const v16h*)&Bs[( 0 + lcol) * 32 + lsel * 16];
    v16h bf1 = *(const v16h*)&Bs[(16 + lcol) * 32 + lsel * 16];
    v16h bf2 = *(const v16h*)&Bs[(32 + lcol) * 32 + lsel * 16];
    v16h bf3 = *(const v16h*)&Bs[(48 + lcol) * 32 + lsel * 16];
    acc0 = __builtin_amdgcn_wmma_f32_16x16x32_f16(false, af, false, bf0, (short)0, acc0, false, false);
    acc1 = __builtin_amdgcn_wmma_f32_16x16x32_f16(false, af, false, bf1, (short)0, acc1, false, false);
    acc2 = __builtin_amdgcn_wmma_f32_16x16x32_f16(false, af, false, bf2, (short)0, acc2, false, false);
    acc3 = __builtin_amdgcn_wmma_f32_16x16x32_f16(false, af, false, bf3, (short)0, acc3, false, false);
  }
  const float* vhb = vh + (size_t)b * NN * DD;
  float* aggb = agg + (size_t)b * NN * DD;
#pragma unroll
  for (int r = 0; r < 8; ++r) {
    int row = m0 + r + 8 * lsel;            // C/D layout: VGPR r -> M=r (lanes<16) / M=r+8
    aggb[row * DD +  0 + lcol] = acc0[r] + vhb[row * DD +  0 + lcol];  // +I self loop
    aggb[row * DD + 16 + lcol] = acc1[r] + vhb[row * DD + 16 + lcol];
    aggb[row * DD + 32 + lcol] = acc2[r] + vhb[row * DD + 32 + lcol];
    aggb[row * DD + 48 + lcol] = acc3[r] + vhb[row * DD + 48 + lcol];
  }
}

// ---------------------------------------------------------------- fused vertex LSTM (WMMA gates + cell)
__global__ void __launch_bounds__(128) k_vlstm(const float* __restrict__ agg,
                                               const float* __restrict__ csum,
                                               float* __restrict__ vh, float* __restrict__ vc,
                                               _Float16* __restrict__ vh16,
                                               const _Float16* __restrict__ Wcat,
                                               const float* __restrict__ bih,
                                               const float* __restrict__ bhh) {
  __shared__ _Float16 Xs[16 * 192];         // [row*192 + k], X = [agg | csum | vh]
  int b = blockIdx.y;
  int m0 = blockIdx.x * 16;
  int wave = threadIdx.x >> 5, lane = threadIdx.x & 31;
  int lcol = lane & 15, lsel = lane >> 4;
  const float* aggb = agg + (size_t)b * NN * DD;
  float* vhb  = vh   + (size_t)b * NN * DD;
  float* vcb  = vc   + (size_t)b * NN * DD;
  _Float16* vh16b = vh16 + (size_t)b * NN * DD;

  for (int i = threadIdx.x; i < 16 * 192; i += 128) {
    int row = i / 192, k = i % 192;
    _Float16 v = (k < 64)  ? (_Float16)aggb[(m0 + row) * DD + k]
               : (k < 128) ? (_Float16)csum[b * DD + (k - 64)]
                           : vh16b[(m0 + row) * DD + (k - 128)];
    Xs[i] = v;
  }
  __syncthreads();

  int dcol = wave * 16 + lcol;
  v8f accI = {}, accF = {}, accG = {}, accO = {};
#pragma unroll
  for (int k0 = 0; k0 < 192; k0 += 32) {
    int kb = k0 + lsel * 16;
    v8h alo = *(const v8h*)&Xs[lcol * 192 + kb];
    v8h ahi = *(const v8h*)&Xs[lcol * 192 + kb + 8];
    v16h af = __builtin_shufflevector(alo, ahi, 0,1,2,3,4,5,6,7,8,9,10,11,12,13,14,15);
    int kk = k0 + lsel * 16;
    v16h wfI = *(const v16h*)&Wcat[(dcol      ) * 192 + kk];
    v16h wfF = *(const v16h*)&Wcat[(dcol +  64) * 192 + kk];
    v16h wfG = *(const v16h*)&Wcat[(dcol + 128) * 192 + kk];
    v16h wfO = *(const v16h*)&Wcat[(dcol + 192) * 192 + kk];
    accI = __builtin_amdgcn_wmma_f32_16x16x32_f16(false, af, false, wfI, (short)0, accI, false, false);
    accF = __builtin_amdgcn_wmma_f32_16x16x32_f16(false, af, false, wfF, (short)0, accF, false, false);
    accG = __builtin_amdgcn_wmma_f32_16x16x32_f16(false, af, false, wfG, (short)0, accG, false, false);
    accO = __builtin_amdgcn_wmma_f32_16x16x32_f16(false, af, false, wfO, (short)0, accO, false, false);
  }

  float bi = bih[dcol]       + bhh[dcol];
  float bf = bih[dcol + 64]  + bhh[dcol + 64];
  float bg = bih[dcol + 128] + bhh[dcol + 128];
  float bo = bih[dcol + 192] + bhh[dcol + 192];
#pragma unroll
  for (int r = 0; r < 8; ++r) {
    int row = m0 + r + 8 * lsel;
    float gi = accI[r] + bi, gf = accF[r] + bf, gg = accG[r] + bg, go = accO[r] + bo;
    float cold = vcb[row * DD + dcol];
    float c2 = sigm(gf) * cold + sigm(gi) * tanhf(gg);
    float h2 = sigm(go) * tanhf(c2);
    vcb[row * DD + dcol]  = c2;
    vhb[row * DD + dcol]  = h2;
    vh16b[row * DD + dcol] = (_Float16)h2;
  }
}

// ---------------------------------------------------------------- v_msg MLP (two passes)
__global__ void k_vmsg1(const float* __restrict__ vh, const float* __restrict__ W1,
                        const float* __restrict__ b1, float* __restrict__ hid) {
  int i = blockIdx.x * blockDim.x + threadIdx.x;   // over B*N*H
  if (i >= BB * NN * HH) return;
  int j = i % HH; int row = i / HH;
  const float* x = vh + (size_t)row * DD;
  const float* w = W1 + j * DD;
  float a = b1[j];
#pragma unroll 8
  for (int k = 0; k < DD; ++k) a += x[k] * w[k];
  hid[i] = fmaxf(a, 0.f);
}

__global__ void k_vmsg2(const float* __restrict__ hid, const float* __restrict__ W2,
                        const float* __restrict__ b2, float* __restrict__ vmsg) {
  int i = blockIdx.x * blockDim.x + threadIdx.x;   // over B*N*D
  if (i >= BB * NN * DD) return;
  int d = i % DD; int row = i / DD;
  const float* h = hid + (size_t)row * HH;
  const float* w = W2 + d * HH;
  float a = b2[d];
#pragma unroll 4
  for (int k = 0; k < HH; ++k) a += h[k] * w[k];
  vmsg[i] = fmaxf(a, 0.f);
}

// ---------------------------------------------------------------- vsum[b,d] = sum_n vmsg  (deterministic tree)
__global__ void k_vsum(const float* __restrict__ vmsg, float* __restrict__ vsum) {
  int b = blockIdx.y, d = blockIdx.x;
  __shared__ float s[256];
  float a = 0.f;
  for (int n = threadIdx.x; n < NN; n += 256) a += vmsg[((size_t)b * NN + n) * DD + d];
  s[threadIdx.x] = a;
  __syncthreads();
  for (int off = 128; off > 0; off >>= 1) {
    if (threadIdx.x < off) s[threadIdx.x] += s[threadIdx.x + off];
    __syncthreads();
  }
  if (threadIdx.x == 0) vsum[b * DD + d] = s[0];
}

// ---------------------------------------------------------------- color LSTM (128 rows)
__global__ void k_clstm(const float* __restrict__ vsum, const int* __restrict__ n_colors,
                        float* __restrict__ ch, float* __restrict__ cc,
                        const float* __restrict__ Wih, const float* __restrict__ Whh,
                        const float* __restrict__ bih, const float* __restrict__ bhh) {
  __shared__ float xin[DD], hrow[DD];
  int bc = blockIdx.x; int b = bc / CC; int c = bc % CC;
  int d = threadIdx.x;                  // 64 threads
  float mask = (c < n_colors[b]) ? 1.f : 0.f;
  xin[d]  = mask * vsum[b * DD + d];
  hrow[d] = ch[bc * DD + d];
  __syncthreads();
  float g[4];
#pragma unroll
  for (int q = 0; q < 4; ++q) {
    int col = q * DD + d;
    const float* wi = Wih + col * DD;
    const float* wh = Whh + col * DD;
    float a = bih[col] + bhh[col];
    for (int k = 0; k < DD; ++k) a += xin[k] * wi[k] + hrow[k] * wh[k];
    g[q] = a;
  }
  float cold = cc[bc * DD + d];
  float c2 = sigm(g[1]) * cold + sigm(g[0]) * tanhf(g[2]);
  float h2 = sigm(g[3]) * tanhf(c2);
  cc[bc * DD + d] = c2;
  ch[bc * DD + d] = h2;
}

// ---------------------------------------------------------------- vote head + mean + sigmoid
__global__ void k_vote(const float* __restrict__ vh,
                       const float* __restrict__ W1, const float* __restrict__ b1,
                       const float* __restrict__ W2, const float* __restrict__ b2,
                       float* __restrict__ out) {
  int b = blockIdx.x;
  __shared__ float s[256];
  float acc = 0.f;
  for (int n = threadIdx.x; n < NN; n += 256) {
    const float* x = vh + ((size_t)b * NN + n) * DD;
    float v = b2[0];
#pragma unroll
    for (int k = 0; k < 16; ++k) {
      const float* w = W1 + k * DD;
      float a = b1[k];
      for (int j = 0; j < DD; ++j) a += x[j] * w[j];
      v += sigm(a) * W2[k];
    }
    acc += v;
  }
  s[threadIdx.x] = acc;
  __syncthreads();
  for (int off = 128; off > 0; off >>= 1) {
    if (threadIdx.x < off) s[threadIdx.x] += s[threadIdx.x + off];
    __syncthreads();
  }
  if (threadIdx.x == 0) out[b] = sigm(s[0] / (float)NN);
}

// ================================================================ host
extern "C" void kernel_launch(void* const* d_in, const int* in_sizes, int n_in,
                              void* d_out, int out_size, void* d_ws, size_t ws_size,
                              hipStream_t stream) {
  const float* Mvv     = (const float*)d_in[0];
  const float* ch_init = (const float*)d_in[1];
  const float* v_init  = (const float*)d_in[2];
  const float* v_Wih   = (const float*)d_in[3];
  const float* v_Whh   = (const float*)d_in[4];
  const float* v_bih   = (const float*)d_in[5];
  const float* v_bhh   = (const float*)d_in[6];
  const float* c_Wih   = (const float*)d_in[7];
  const float* c_Whh   = (const float*)d_in[8];
  const float* c_bih   = (const float*)d_in[9];
  const float* c_bhh   = (const float*)d_in[10];
  const float* cmsg_W1 = (const float*)d_in[11];
  const float* cmsg_b1 = (const float*)d_in[12];
  const float* cmsg_W2 = (const float*)d_in[13];
  const float* cmsg_b2 = (const float*)d_in[14];
  const float* vmsg_W1 = (const float*)d_in[15];
  const float* vmsg_b1 = (const float*)d_in[16];
  const float* vmsg_W2 = (const float*)d_in[17];
  const float* vmsg_b2 = (const float*)d_in[18];
  const float* vote_W1 = (const float*)d_in[19];
  const float* vote_b1 = (const float*)d_in[20];
  const float* vote_W2 = (const float*)d_in[21];
  const float* vote_b2 = (const float*)d_in[22];
  const int*   n_col   = (const int*)d_in[23];

  const size_t SZ_ND = (size_t)BB * NN * DD;   // 1,048,576
  const size_t SZ_NH = (size_t)BB * NN * HH;   // 1,638,400
  const size_t SZ_CD = (size_t)BB * CC * DD;   // 8,192
  const size_t SZ_D  = (size_t)BB * DD;        // 512
  const size_t SZ_A  = (size_t)BB * NN * NN;   // 33,554,432

  float* p    = (float*)d_ws;
  float* vh   = p;             p += SZ_ND;
  float* vc   = p;             p += SZ_ND;
  float* agg  = p;             p += SZ_ND;
  float* vmsg = p;             p += SZ_ND;
  float* hid  = p;             p += SZ_NH;
  float* cmsg = p;             p += SZ_CD;
  float* csum = p;             p += SZ_D;
  float* ch   = p;             p += SZ_CD;
  float* cc   = p;             p += SZ_CD;
  float* vsum = p;             p += SZ_D;
  _Float16* h16  = (_Float16*)p;
  _Float16* Ah   = h16;                         // 64 MB
  _Float16* vh16 = Ah + SZ_A;                   // 2 MB
  _Float16* Wcat = vh16 + SZ_ND;                // 96 KB

  k_init<<<(int)((SZ_ND + 255) / 256), 256, 0, stream>>>(ch_init, v_init, vh, vc, ch, cc, vh16);
  k_convA<<<(int)((SZ_A / 4 + 255) / 256), 256, 0, stream>>>(Mvv, Ah);
  k_convW<<<(256 * 192 + 255) / 256, 256, 0, stream>>>(v_Wih, v_Whh, Wcat);

  for (int t = 0; t < TT; ++t) {
    k_cmsg<<<BB * CC, 128, 0, stream>>>(ch, cmsg_W1, cmsg_b1, cmsg_W2, cmsg_b2, cmsg);
    k_csum<<<BB, DD, 0, stream>>>(cmsg, n_col, csum);
    k_agg<<<dim3(NN / 64, BB), 128, 0, stream>>>(Ah, vh16, vh, agg);
    k_vlstm<<<dim3(NN / 16, BB), 128, 0, stream>>>(agg, csum, vh, vc, vh16, Wcat, v_bih, v_bhh);
    k_vmsg1<<<(int)((SZ_NH + 255) / 256), 256, 0, stream>>>(vh, vmsg_W1, vmsg_b1, hid);
    k_vmsg2<<<(int)((SZ_ND + 255) / 256), 256, 0, stream>>>(hid, vmsg_W2, vmsg_b2, vmsg);
    k_vsum<<<dim3(DD, BB), 256, 0, stream>>>(vmsg, vsum);
    k_clstm<<<BB * CC, DD, 0, stream>>>(vsum, n_col, ch, cc, c_Wih, c_Whh, c_bih, c_bhh);
  }

  k_vote<<<BB, 256, 0, stream>>>(vh, vote_W1, vote_b1, vote_W2, vote_b2, (float*)d_out);
}
